// Attention_652835029662
// MI455X (gfx1250) — matrix operand order repaired and checked by reading, not compiled, not run on hardware
//
#include <hip/hip_runtime.h>

typedef __bf16 bf16;
typedef __attribute__((ext_vector_type(16))) __bf16 v16bf;
typedef __attribute__((ext_vector_type(8)))  __bf16 v8bf;
typedef __attribute__((ext_vector_type(8)))  float  v8f;
typedef __attribute__((ext_vector_type(4)))  int    v4i;

#define B_     8
#define N_     1024
#define DIM_   512
#define HEADS_ 8
#define DHEAD_ 64
#define INNER_ 512
#define HB_    (HEADS_ * B_)     // 64
#define ROWS_  (B_ * N_)         // 8192

#ifndef __has_builtin
#define __has_builtin(x) 0
#endif
#if defined(__HIP_DEVICE_COMPILE__) &&                                        \
    __has_builtin(__builtin_amdgcn_global_load_async_to_lds_b128) &&          \
    __has_builtin(__builtin_amdgcn_s_wait_asynccnt)
#define HAVE_ASYNC_LDS 1
#else
#define HAVE_ASYNC_LDS 0
#endif

#define AS1 __attribute__((address_space(1)))
#define AS3 __attribute__((address_space(3)))

// 16-byte global -> LDS copy. Async (ASYNCcnt-tracked, no VGPR bounce) when the
// gfx1250 async builtins are available; plain load/store otherwise.
__device__ __forceinline__ void cp16_g2l(bf16* ldst, const bf16* gsrc) {
#if HAVE_ASYNC_LDS
  __builtin_amdgcn_global_load_async_to_lds_b128((AS1 v4i*)gsrc,
                                                 (AS3 v4i*)ldst, 0, 0);
#else
  *(v8bf*)ldst = *(const v8bf*)gsrc;
#endif
}
__device__ __forceinline__ void wait_async(int pending_is_2) {
#if HAVE_ASYNC_LDS
  if (pending_is_2)
    __builtin_amdgcn_s_wait_asynccnt(2);
  else
    __builtin_amdgcn_s_wait_asynccnt(0);
#else
  (void)pending_is_2;
#endif
}

// ---------------- WMMA helpers (CDNA5 16x16x32 bf16, f32 accum) ----------------
// A-frag (16x32, M x K): lane L -> row m = L&15, hi = L>>4.
//   a[0..7]  = A[m][kbase + 8*hi + i]
//   a[8..15] = A[m][kbase + 16 + 8*hi + i]
__device__ __forceinline__ v16bf afrag(const bf16* row, int kbase, int hi) {
  v8bf c0 = *(const v8bf*)(row + kbase + 8 * hi);
  v8bf c1 = *(const v8bf*)(row + kbase + 16 + 8 * hi);
  v16bf a;
#pragma unroll
  for (int i = 0; i < 8; ++i) { a[i] = c0[i]; a[i + 8] = c1[i]; }
  return a;
}
// B-frag (32x16, K x N) read from row-major B^T: lane L -> col n = L&15, khi = L>>4.
//   b[0..7] = Bt[n][kbase + 8*khi + i], b[8..15] = Bt[n][kbase + 16 + 8*khi + i]
__device__ __forceinline__ v16bf bfrag(const bf16* btrow, int kbase, int khi) {
  v8bf c0 = *(const v8bf*)(btrow + kbase + 8 * khi);
  v8bf c1 = *(const v8bf*)(btrow + kbase + 16 + 8 * khi);
  v16bf b;
#pragma unroll
  for (int i = 0; i < 8; ++i) { b[i] = c0[i]; b[i + 8] = c1[i]; }
  return b;
}
__device__ __forceinline__ v8f wmma_bf16(v16bf a, v16bf b, v8f c) {
  return __builtin_amdgcn_wmma_f32_16x16x32_bf16(false, a, false, b, (short)0, c,
                                                 false, false);
}

// ---------------- Kernel 0: fp32 -> bf16 weight conversion ----------------
__global__ void k_cvt(const float* __restrict__ src, bf16* __restrict__ dst, int n) {
  int i = blockIdx.x * blockDim.x + threadIdx.x;
  if (i < n) dst[i] = (bf16)src[i];
}

// ---------------- Kernel 1: LayerNorm + input projection ----------------
// Block: 256 threads = 8 waves, handles 16 rows of one of {q,k,v}.
// Each wave: LN for 2 rows, then WMMA for 4 o-tiles (16 cols each) == one head.
__global__ __launch_bounds__(256) void k_lnproj(
    const float* __restrict__ q, const float* __restrict__ k,
    const float* __restrict__ v, const float* __restrict__ g,
    const float* __restrict__ beta, const bf16* __restrict__ winb,
    bf16* __restrict__ fQ, bf16* __restrict__ fK, bf16* __restrict__ vT) {
  __shared__ __align__(16) bf16 lA[16][DIM_ + 8];  // 16B-aligned rows
  const int t  = blockIdx.x / (ROWS_ / 16);
  const int rt = blockIdx.x % (ROWS_ / 16);
  const int r0 = rt * 16;
  const float* x = (t == 0) ? q : (t == 1) ? k : v;
  const int w = threadIdx.x >> 5, lane = threadIdx.x & 31;

  // --- LayerNorm: wave w does local rows w*2, w*2+1 ---
#pragma unroll
  for (int rr = 0; rr < 2; ++rr) {
    const int rl = w * 2 + rr;
    const float* xr = x + (size_t)(r0 + rl) * DIM_;
    float xs[16], s = 0.f, s2 = 0.f;
#pragma unroll
    for (int j = 0; j < 16; ++j) {
      float xv = xr[lane + 32 * j];
      xs[j] = xv; s += xv; s2 += xv * xv;
    }
#pragma unroll
    for (int m = 16; m >= 1; m >>= 1) { s += __shfl_xor(s, m); s2 += __shfl_xor(s2, m); }
    const float mu = s * (1.f / DIM_);
    const float rs = rsqrtf(s2 * (1.f / DIM_) - mu * mu + 1e-5f);
#pragma unroll
    for (int j = 0; j < 16; ++j) {
      const int c = lane + 32 * j;
      lA[rl][c] = (bf16)((xs[j] - mu) * rs * g[c] + beta[c]);
    }
  }
  __syncthreads();

  // --- GEMM tile: D[16 x 16] per o-tile, K = 512 ---
  const int hi = lane >> 4, col = lane & 15;
  const bf16* arow = &lA[col][0];
#pragma unroll
  for (int ot = 0; ot < 4; ++ot) {
    const int o0 = (w * 4 + ot) * 16;                    // wave w owns head w
    const bf16* brow = winb + (size_t)(o0 + col) * DIM_; // W_in row-major == B^T
    v8f acc = {};
#pragma unroll
    for (int kb = 0; kb < DIM_; kb += 32)
      acc = wmma_bf16(afrag(arow, kb, hi), bfrag(brow, kb, hi), acc);

    const int h = o0 >> 6;
    const int d = (o0 & 63) + col;
#pragma unroll
    for (int vv = 0; vv < 8; ++vv) {
      const int rg = r0 + vv + 8 * hi;          // global row in [0, 8192)
      const int b_ = rg >> 10, n = rg & (N_ - 1);
      const float val = acc[vv];
      if (t == 0)
        fQ[(((size_t)h * B_ + b_) * N_ + n) * DHEAD_ + d] = (bf16)val;
      else if (t == 1)
        fK[(((size_t)h * B_ + b_) * N_ + n) * DHEAD_ + d] = (bf16)val;
      else  // store V transposed: vT[h][b][d][n]
        vT[(((size_t)h * B_ + b_) * DHEAD_ + d) * N_ + n] = (bf16)val;
    }
  }
}

// ---------------- Kernel 2: per-row inv-norm and mean over head dim ----------------
// invn/mean layout: [2][HB][N]  (slot 0 = q, slot 1 = k)
__global__ void k_stats(const bf16* __restrict__ fQ, const bf16* __restrict__ fK,
                        float* __restrict__ invn, float* __restrict__ mean) {
  const int i = blockIdx.x * blockDim.x + threadIdx.x;
  const int total = 2 * HB_ * N_;
  if (i >= total) return;
  const int tt = i / (HB_ * N_);
  const int rem = i % (HB_ * N_);
  const bf16* p = (tt ? fK : fQ) + (size_t)rem * DHEAD_;
  float s = 0.f, s2 = 0.f;
#pragma unroll
  for (int j = 0; j < DHEAD_; ++j) { float xv = (float)p[j]; s += xv; s2 += xv * xv; }
  invn[i] = rsqrtf(s2 + 1e-30f);
  mean[i] = s * (1.f / DHEAD_);
}

// ---------------- Kernel 3: attention (no softmax; affine-rescaled dots) ----------
// grid = (64 hb, 8 q-tiles). Block 256 = 8 waves; wave owns 16 q rows.
// scores = dots*(0.7*invq*invk + 0.3/64) - 0.3*qm*km   (cov folded into dots)
// K/V tiles double-buffered in LDS via async global->LDS copies (ASYNCcnt).
__global__ __launch_bounds__(256) void k_attn(
    const bf16* __restrict__ fQ, const bf16* __restrict__ fK,
    const bf16* __restrict__ vT, const float* __restrict__ invn,
    const float* __restrict__ mean, bf16* __restrict__ attn) {
  __shared__ __align__(16) bf16 lK[2][32][72];    // K-chunk rows (B^T for dots)
  __shared__ __align__(16) bf16 lV[2][64][40];    // vT slice (B^T for out)
  __shared__ __align__(16) bf16 lS[8][16][40];    // per-wave score tile 16x32

  const int hb = blockIdx.x;                       // h*B + b
  const int h = hb >> 3, b_ = hb & 7;
  const int w = threadIdx.x >> 5, lane = threadIdx.x & 31;
  const int hi = lane >> 4, col = lane & 15;
  const int q0 = blockIdx.y * 128 + w * 16;

  // per-thread staging coordinates (2048 bf16 per tile, 8 per thread)
  const int e  = threadIdx.x * 8;
  const int mr = e >> 6, cc = e & 63;              // K tile: row, col
  const int dv = threadIdx.x >> 2, m8 = (threadIdx.x & 3) * 8;  // V tile
  const bf16* gK = fK + (size_t)hb * N_ * DHEAD_;
  const bf16* gV = vT + (size_t)hb * DHEAD_ * N_;

  auto fill = [&](int bi, int mc) {
    cp16_g2l(&lK[bi][mr][cc], gK + (size_t)(mc + mr) * DHEAD_ + cc);
    cp16_g2l(&lV[bi][dv][m8], gV + (size_t)dv * N_ + mc + m8);
  };

  const bf16* qrow = fQ + ((size_t)hb * N_ + q0 + col) * DHEAD_;
  const v16bf aq0 = afrag(qrow, 0, hi);
  const v16bf aq1 = afrag(qrow, 32, hi);

  const float* invq = invn + (size_t)hb * N_;
  const float* qm   = mean + (size_t)hb * N_;
  const float* invk = invn + (size_t)(HB_ + hb) * N_;
  const float* km   = mean + (size_t)(HB_ + hb) * N_;
  float invq_r[8], qm_r[8];
#pragma unroll
  for (int vv = 0; vv < 8; ++vv) {
    const int n = q0 + vv + 8 * hi;
    invq_r[vv] = invq[n];
    qm_r[vv]   = qm[n];
  }

  v8f o0 = {}, o1 = {}, o2 = {}, o3 = {};

  const int T = N_ / 32;                           // 32 chunks
  fill(0, 0);                                      // prologue
  for (int it = 0; it < T; ++it) {
    const int cur = it & 1;
    const int mc = it * 32;
    const int more = (it + 1 < T);
    if (more) fill(cur ^ 1, mc + 32);              // prefetch next chunk
    wait_async(more);                              // cur buffer resident
    __syncthreads();

    // ---- dots for two 16-wide key sub-tiles, then affine -> bf16 S tile ----
#pragma unroll
    for (int sub = 0; sub < 2; ++sub) {
      const bf16* krow = &lK[cur][sub * 16 + col][0];
      v8f dacc = {};
      dacc = wmma_bf16(aq0, bfrag(krow, 0, hi), dacc);
      dacc = wmma_bf16(aq1, bfrag(krow, 32, hi), dacc);
      const int mglob = mc + sub * 16 + col;
      const float invk_m = invk[mglob];
      const float km_m   = km[mglob];
#pragma unroll
      for (int vv = 0; vv < 8; ++vv) {
        const float sc = dacc[vv] * (0.7f * invq_r[vv] * invk_m + 0.3f / 64.f) -
                         0.3f * qm_r[vv] * km_m;
        lS[w][vv + 8 * hi][sub * 16 + col] = (bf16)sc;
      }
    }

    // ---- O += S(16x32) x V(32x64): 4 d-tiles ----
    {
      const bf16* srow = &lS[w][col][0];
      const v16bf as = afrag(srow, 0, hi);
      o0 = wmma_bf16(as, bfrag(&lV[cur][ 0 + col][0], 0, hi), o0);
      o1 = wmma_bf16(as, bfrag(&lV[cur][16 + col][0], 0, hi), o1);
      o2 = wmma_bf16(as, bfrag(&lV[cur][32 + col][0], 0, hi), o2);
      o3 = wmma_bf16(as, bfrag(&lV[cur][48 + col][0], 0, hi), o3);
    }
    __syncthreads();   // everyone done reading cur before it is refilled
  }

  // ---- store O as bf16 into attn[b][n][h*64 + d] ----
#pragma unroll
  for (int vv = 0; vv < 8; ++vv) {
    const int n = q0 + vv + 8 * hi;
    const size_t base = ((size_t)b_ * N_ + n) * INNER_ + h * DHEAD_;
    attn[base +  0 + col] = (bf16)o0[vv];
    attn[base + 16 + col] = (bf16)o1[vv];
    attn[base + 32 + col] = (bf16)o2[vv];
    attn[base + 48 + col] = (bf16)o3[vv];
  }
}

// ---------------- Kernel 4: output projection + bias (fp32 out) ----------------
__global__ __launch_bounds__(256) void k_outproj(
    const bf16* __restrict__ attn, const bf16* __restrict__ woutb,
    const float* __restrict__ bias, float* __restrict__ out) {
  __shared__ __align__(16) bf16 lA[16][DIM_ + 8];
  const int r0 = blockIdx.x * 16;
  const int w = threadIdx.x >> 5, lane = threadIdx.x & 31;
  {
    const int row = threadIdx.x >> 4;
    const int c0 = (threadIdx.x & 15) * 32;
    const bf16* src = attn + (size_t)(r0 + row) * INNER_ + c0;
#pragma unroll
    for (int j = 0; j < 4; ++j)
      cp16_g2l(&lA[row][c0 + j * 8], src + j * 8);
    wait_async(0);
  }
  __syncthreads();

  const int hi = lane >> 4, col = lane & 15;
  const bf16* arow = &lA[col][0];
#pragma unroll
  for (int ot = 0; ot < 4; ++ot) {
    const int o0 = (w * 4 + ot) * 16;
    const bf16* brow = woutb + (size_t)(o0 + col) * INNER_;  // W_out row-major == B^T
    v8f acc = {};
#pragma unroll
    for (int kb = 0; kb < INNER_; kb += 32)
      acc = wmma_bf16(afrag(arow, kb, hi), bfrag(brow, kb, hi), acc);
    const float bo = bias[o0 + col];
#pragma unroll
    for (int vv = 0; vv < 8; ++vv)
      out[(size_t)(r0 + vv + 8 * hi) * DIM_ + o0 + col] = acc[vv] + bo;
  }
}

// ---------------- launch ----------------
extern "C" void kernel_launch(void* const* d_in, const int* in_sizes, int n_in,
                              void* d_out, int out_size, void* d_ws, size_t ws_size,
                              hipStream_t stream) {
  (void)in_sizes; (void)n_in; (void)out_size; (void)ws_size;
  const float* q    = (const float*)d_in[0];
  const float* k    = (const float*)d_in[1];
  const float* v    = (const float*)d_in[2];
  const float* g    = (const float*)d_in[3];
  const float* beta = (const float*)d_in[4];
  const float* win  = (const float*)d_in[5];
  const float* wout = (const float*)d_in[6];
  const float* bout = (const float*)d_in[7];
  float* out = (float*)d_out;

  char* ws = (char*)d_ws;
  size_t off = 0;
  auto alloc = [&](size_t bytes) -> void* {
    void* p = ws + off;
    off = (off + bytes + 255) & ~(size_t)255;
    return p;
  };
  const size_t fElems = (size_t)HB_ * N_ * DHEAD_;        // 4 Mi
  bf16* fQ    = (bf16*)alloc(fElems * 2);
  bf16* fK    = (bf16*)alloc(fElems * 2);
  bf16* vT    = (bf16*)alloc(fElems * 2);
  bf16* attn  = (bf16*)alloc((size_t)ROWS_ * INNER_ * 2);
  bf16* winb  = (bf16*)alloc((size_t)INNER_ * DIM_ * 2);
  bf16* woutb = (bf16*)alloc((size_t)DIM_ * INNER_ * 2);
  float* invn = (float*)alloc((size_t)2 * HB_ * N_ * 4);
  float* mean = (float*)alloc((size_t)2 * HB_ * N_ * 4);

  const int nW = INNER_ * DIM_;  // 262144
  k_cvt<<<(nW + 255) / 256, 256, 0, stream>>>(win, winb, nW);
  k_cvt<<<(nW + 255) / 256, 256, 0, stream>>>(wout, woutb, nW);

  k_lnproj<<<3 * (ROWS_ / 16), 256, 0, stream>>>(q, k, v, g, beta, winb, fQ, fK, vT);

  k_stats<<<(2 * HB_ * N_) / 256, 256, 0, stream>>>(fQ, fK, invn, mean);

  dim3 ag(HB_, N_ / 128);
  k_attn<<<ag, 256, 0, stream>>>(fQ, fK, vT, invn, mean, attn);

  k_outproj<<<ROWS_ / 16, 256, 0, stream>>>(attn, woutb, bout, out);
}
